// Attention_48868137893979
// MI455X (gfx1250) — matrix operand order repaired and checked by reading, not compiled, not run on hardware
//
#include <hip/hip_runtime.h>

// ---------------------------------------------------------------------------
// MI455X (gfx1250) attention layer: f16 WMMA GEMMs + WMMA flash attention.
// All matrix math goes through v_wmma_f32_16x16x32_f16 (f32 accumulation).
// Wave tiles sized for loads-per-WMMA ~1.5 (32xM tiles reuse B fragments).
// ---------------------------------------------------------------------------

typedef __attribute__((ext_vector_type(16))) _Float16 v16h;
typedef __attribute__((ext_vector_type(8)))  _Float16 v8h;
typedef __attribute__((ext_vector_type(4)))  _Float16 v4h;
typedef __attribute__((ext_vector_type(8)))  float    v8f;
typedef __attribute__((ext_vector_type(4)))  float    v4f;

namespace {
constexpr int kB    = 2;
constexpr int kL    = 1024;
constexpr int kP    = 4;
constexpr int kHID  = 1024;
constexpr int kNH   = 16;
constexpr int kHS   = 64;
constexpr int kTOK  = kB * kL * kP;    // 8192 tokens
constexpr int kD    = kNH * kHS;       // 1024
constexpr int kSLAB = kB * kNH * kP;   // 128 attention slabs
}

union H16 { v16h v; v8h h[2]; };

// --- A fragment (16x32, f16). Row-major src, `ld` elements per row.
// Lane layout (ISA 7.12.2): lanes 0-15 row M=lane, K={0..7,16..23};
// lanes 16-31 row M=lane-16, K={8..15,24..31}.
__device__ __forceinline__ v16h load_fragA(const _Float16* base, int ld, int lane) {
  const int r  = lane & 15;
  const int hh = (lane >> 4) & 1;
  const _Float16* p = base + (size_t)r * ld + hh * 8;
  H16 f;
  f.h[0] = *(const v8h*)(p);        // K = hh*8 + 0..7
  f.h[1] = *(const v8h*)(p + 16);   // K = 16 + hh*8 + 0..7
  return f.v;
}

// --- B fragment (32x16, f16). Column n = (lane&15) is a *row* of the
// row-major source (i.e. source holds B^T, as for W[d][k] or V^T[hs][l]).
// Lanes 0-15 hold K={0..7,16..23}, lanes 16-31 hold K={8..15,24..31} (matches A).
__device__ __forceinline__ v16h load_fragB(const _Float16* base, int ld, int lane) {
  const int n  = lane & 15;
  const int hh = (lane >> 4) & 1;
  const _Float16* p = base + (size_t)n * ld + hh * 8;
  H16 f;
  f.h[0] = *(const v8h*)(p);
  f.h[1] = *(const v8h*)(p + 16);
  return f.v;
}

// ---------------------------------------------------------------------------
// fp32 -> f16 conversion, 4 elements/thread (b128 in, b64 out)
// ---------------------------------------------------------------------------
__global__ void cvt_f32_f16_v4(const float* __restrict__ src,
                               _Float16* __restrict__ dst, int n4) {
  int i = blockIdx.x * blockDim.x + threadIdx.x;
  if (i < n4) {
    v4f f = ((const v4f*)src)[i];
    ((v4h*)dst)[i] = __builtin_convertvector(f, v4h);
  }
}

// ---------------------------------------------------------------------------
// Projection GEMM: dst[token,d] = sum_k X[token,k] * W[d,k]
// Wave tile: 32 tokens x 64 out-dims (8 WMMA / 12 b128-loads per K-step).
// Output scattered to slab layout [B, NH, P, L, HS] as f16.
// ---------------------------------------------------------------------------
__global__ void __launch_bounds__(256)
proj_wmma(const _Float16* __restrict__ X, const _Float16* __restrict__ W,
          _Float16* __restrict__ dst) {
  const int lane = threadIdx.x & 31;
  const int wid  = threadIdx.x >> 5;
  const int task = blockIdx.x * 8 + wid;     // 4096 tasks
  const int m0   = (task >> 4) * 32;         // 256 token tiles of 32
  const int n0   = (task & 15) * 64;         // 16 output super-tiles

  v8f z = {};
  v8f acc[2][4];
#pragma unroll
  for (int rb = 0; rb < 2; ++rb)
#pragma unroll
    for (int t = 0; t < 4; ++t) acc[rb][t] = z;

  const _Float16* A0 = X + (size_t)m0 * kHID;
  const _Float16* A1 = X + (size_t)(m0 + 16) * kHID;
  for (int k0 = 0; k0 < kHID; k0 += 32) {
    __builtin_prefetch(A0 + k0 + 256, 0, 3);   // global_prefetch_b8
    v16h a0 = load_fragA(A0 + k0, kHID, lane);
    v16h a1 = load_fragA(A1 + k0, kHID, lane);
#pragma unroll
    for (int t = 0; t < 4; ++t) {
      v16h bt = load_fragB(W + (size_t)(n0 + t * 16) * kHID + k0, kHID, lane);
      acc[0][t] = __builtin_amdgcn_wmma_f32_16x16x32_f16(
          false, a0, false, bt, (short)0, acc[0][t], false, false);
      acc[1][t] = __builtin_amdgcn_wmma_f32_16x16x32_f16(
          false, a1, false, bt, (short)0, acc[1][t], false, false);
    }
  }

  const int hh = (lane >> 4) & 1;
  const int c  = lane & 15;
#pragma unroll
  for (int rb = 0; rb < 2; ++rb) {
#pragma unroll
    for (int t = 0; t < 4; ++t) {
#pragma unroll
      for (int j = 0; j < 8; ++j) {
        const int tok = m0 + rb * 16 + j + 8 * hh;   // C layout: row=j+8*half
        const int d   = n0 + t * 16 + c;
        const int pp  = tok & 3;
        const int l   = (tok >> 2) & (kL - 1);
        const int bb  = tok >> 12;
        const int n   = d >> 6;
        const int hs  = d & 63;
        const size_t idx =
            ((((size_t)bb * kNH + n) * kP + pp) * kL + l) * kHS + hs;
        dst[idx] = (_Float16)acc[rb][t][j];
      }
    }
  }
}

// ---------------------------------------------------------------------------
// RoPE, in place on f16 slab tensor [128, L, HS]; one thread per (s,l,d<32).
// cos/sin tables repeat at d and d+32 (emb = concat(freqs,freqs)).
// ---------------------------------------------------------------------------
__global__ void rope_inplace(_Float16* __restrict__ t,
                             const float* __restrict__ cosT,
                             const float* __restrict__ sinT) {
  int i = blockIdx.x * blockDim.x + threadIdx.x;   // 128*1024*32
  if (i >= kSLAB * kL * 32) return;
  const int d = i & 31;
  const int l = (i >> 5) & (kL - 1);
  const int s = i >> 15;
  const size_t base = ((size_t)s * kL + l) * kHS;
  const float c  = cosT[l * kHS + d];
  const float sn = sinT[l * kHS + d];
  const float t1 = (float)t[base + d];
  const float t2 = (float)t[base + d + 32];
  t[base + d]      = (_Float16)(t1 * c - t2 * sn);
  t[base + d + 32] = (_Float16)(t2 * c + t1 * sn);
}

// ---------------------------------------------------------------------------
// V transpose: [128, L, HS] -> [128, HS, L] so PV B-fragments are contiguous.
// ---------------------------------------------------------------------------
__global__ void transpose_v(const _Float16* __restrict__ v,
                            _Float16* __restrict__ vt) {
  int i = blockIdx.x * blockDim.x + threadIdx.x;   // 128*64*1024
  if (i >= kSLAB * kHS * kL) return;
  const int l  = i & (kL - 1);
  const int hs = (i >> 10) & 63;
  const int s  = i >> 16;
  vt[i] = v[((size_t)s * kL + l) * kHS + hs];      // vt[s][hs][l]
}

// ---------------------------------------------------------------------------
// Flash attention (causal), one wave per (slab, 32-query block): every K/V
// fragment is reused by two 16-row blocks. Q.K^T and P.V via WMMA f16;
// online softmax in fp32; P re-layout via per-wave LDS tile + s_wait_dscnt.
// Output written as f16 to token-major [TOK, D] for the output GEMM.
// ---------------------------------------------------------------------------
__global__ void __launch_bounds__(256)
flash_wmma(const _Float16* __restrict__ Q, const _Float16* __restrict__ K,
           const _Float16* __restrict__ VT, _Float16* __restrict__ O) {
  __shared__ __align__(16) _Float16 sP[8][2][16][32];   // 2 KB per wave

  const int lane = threadIdx.x & 31;
  const int wid  = threadIdx.x >> 5;
  const int task = blockIdx.x * 8 + wid;   // 4096 = 128 slabs * 32 qblocks
  const int s    = task >> 5;
  const int q0   = (task & 31) * 32;

  const _Float16* qs = Q + (size_t)s * kL * kHS;
  const _Float16* ks = K + (size_t)s * kL * kHS;
  const _Float16* vs = VT + (size_t)s * kHS * kL;

  const int hh = (lane >> 4) & 1;
  const int c  = lane & 15;

  // Q fragments: 2 row blocks x 2 K-chunks, resident for the whole key loop.
  v16h aq[2][2];
#pragma unroll
  for (int rb = 0; rb < 2; ++rb)
#pragma unroll
    for (int ch = 0; ch < 2; ++ch)
      aq[rb][ch] =
          load_fragA(qs + (size_t)(q0 + rb * 16) * kHS + ch * 32, kHS, lane);

  v8f z = {};
  v8f acc[2][4];
  float mrow[2][8], lrow[2][8];
#pragma unroll
  for (int rb = 0; rb < 2; ++rb) {
#pragma unroll
    for (int t = 0; t < 4; ++t) acc[rb][t] = z;
#pragma unroll
    for (int j = 0; j < 8; ++j) {
      mrow[rb][j] = -__builtin_inff();
      lrow[rb][j] = 0.f;
    }
  }

  const float scale = 0.125f;   // 1/sqrt(64)

  for (int k0 = 0; k0 <= q0 + 31; k0 += 32) {
    // ---- S = Q.K^T : two key tiles, shared across both row blocks ----
    v8f sc[2][2];
#pragma unroll
    for (int t = 0; t < 2; ++t) {
      v16h b0 = load_fragB(ks + (size_t)(k0 + t * 16) * kHS + 0,  kHS, lane);
      v16h b1 = load_fragB(ks + (size_t)(k0 + t * 16) * kHS + 32, kHS, lane);
#pragma unroll
      for (int rb = 0; rb < 2; ++rb) {
        sc[rb][t] = __builtin_amdgcn_wmma_f32_16x16x32_f16(
            false, aq[rb][0], false, b0, (short)0, z, false, false);
        sc[rb][t] = __builtin_amdgcn_wmma_f32_16x16x32_f16(
            false, aq[rb][1], false, b1, (short)0, sc[rb][t], false, false);
      }
    }

#pragma unroll
    for (int rb = 0; rb < 2; ++rb) {
      // ---- scale + causal mask (row = q0+rb*16+j+8*hh, col = k0+t*16+c) ----
#pragma unroll
      for (int t = 0; t < 2; ++t) {
        const int kidx = k0 + t * 16 + c;
#pragma unroll
        for (int j = 0; j < 8; ++j) {
          const int r = q0 + rb * 16 + j + 8 * hh;
          const float sv = sc[rb][t][j] * scale;
          sc[rb][t][j] = (kidx <= r) ? sv : -__builtin_inff();
        }
      }
      // ---- online softmax: row max/sum across the 16-lane group ----
      float fj[8];
#pragma unroll
      for (int j = 0; j < 8; ++j) {
        float mx = fmaxf(sc[rb][0][j], sc[rb][1][j]);
#pragma unroll
        for (int m = 8; m >= 1; m >>= 1) mx = fmaxf(mx, __shfl_xor(mx, m, 32));
        const float mnew = fmaxf(mrow[rb][j], mx);   // finite after k0=0
        fj[j]       = __expf(mrow[rb][j] - mnew);
        mrow[rb][j] = mnew;
        const float p0 = __expf(sc[rb][0][j] - mnew);
        const float p1 = __expf(sc[rb][1][j] - mnew);
        sc[rb][0][j] = p0;
        sc[rb][1][j] = p1;
        float rs = p0 + p1;
#pragma unroll
        for (int m = 8; m >= 1; m >>= 1) rs += __shfl_xor(rs, m, 32);
        lrow[rb][j] = lrow[rb][j] * fj[j] + rs;
      }
      // ---- rescale accumulators ----
#pragma unroll
      for (int t = 0; t < 4; ++t)
#pragma unroll
        for (int j = 0; j < 8; ++j) acc[rb][t][j] *= fj[j];
      // ---- stage P (C layout) into LDS ----
#pragma unroll
      for (int t = 0; t < 2; ++t)
#pragma unroll
        for (int j = 0; j < 8; ++j)
          sP[wid][rb][j + 8 * hh][t * 16 + c] = (_Float16)sc[rb][t][j];
    }

    asm volatile("s_wait_dscnt 0" ::: "memory");

    // ---- reload P as A fragments (16x32 each) ----
    H16 pf[2];
#pragma unroll
    for (int rb = 0; rb < 2; ++rb) {
      const _Float16* pp = &sP[wid][rb][lane & 15][hh * 8];
      pf[rb].h[0] = *(const v8h*)(pp);
      pf[rb].h[1] = *(const v8h*)(pp + 16);
    }

    // ---- O += P.V : V^T B fragments shared across both row blocks ----
#pragma unroll
    for (int nt = 0; nt < 4; ++nt) {
      v16h bv = load_fragB(vs + (size_t)(nt * 16) * kL + k0, kL, lane);
      acc[0][nt] = __builtin_amdgcn_wmma_f32_16x16x32_f16(
          false, pf[0].v, false, bv, (short)0, acc[0][nt], false, false);
      acc[1][nt] = __builtin_amdgcn_wmma_f32_16x16x32_f16(
          false, pf[1].v, false, bv, (short)0, acc[1][nt], false, false);
    }
  }

  // ---- epilogue: normalize and scatter to token-major [TOK, D] ----
  const int bb = s >> 6;
  const int n  = (s >> 2) & 15;
  const int pp = s & 3;
#pragma unroll
  for (int rb = 0; rb < 2; ++rb) {
#pragma unroll
    for (int j = 0; j < 8; ++j) {
      const int r = q0 + rb * 16 + j + 8 * hh;        // query position l
      const size_t tok = ((size_t)bb * kL + r) * kP + pp;
      const float inv = 1.0f / lrow[rb][j];
#pragma unroll
      for (int nt = 0; nt < 4; ++nt) {
        const int d = n * kHS + nt * 16 + c;
        O[tok * kD + d] = (_Float16)(acc[rb][nt][j] * inv);
      }
    }
  }
}

// ---------------------------------------------------------------------------
// Output projection: out[token,hid] = sum_d O[token,d] * OW[hid,d], fp32 out.
// Wave tile 32x64, same structure as proj_wmma.
// ---------------------------------------------------------------------------
__global__ void __launch_bounds__(256)
oproj_wmma(const _Float16* __restrict__ Ob, const _Float16* __restrict__ W,
           float* __restrict__ out) {
  const int lane = threadIdx.x & 31;
  const int wid  = threadIdx.x >> 5;
  const int task = blockIdx.x * 8 + wid;     // 4096 tasks
  const int m0   = (task >> 4) * 32;
  const int n0   = (task & 15) * 64;

  v8f z = {};
  v8f acc[2][4];
#pragma unroll
  for (int rb = 0; rb < 2; ++rb)
#pragma unroll
    for (int t = 0; t < 4; ++t) acc[rb][t] = z;

  const _Float16* A0 = Ob + (size_t)m0 * kD;
  const _Float16* A1 = Ob + (size_t)(m0 + 16) * kD;
  for (int k0 = 0; k0 < kD; k0 += 32) {
    __builtin_prefetch(A0 + k0 + 256, 0, 3);
    v16h a0 = load_fragA(A0 + k0, kD, lane);
    v16h a1 = load_fragA(A1 + k0, kD, lane);
#pragma unroll
    for (int t = 0; t < 4; ++t) {
      v16h bt = load_fragB(W + (size_t)(n0 + t * 16) * kD + k0, kD, lane);
      acc[0][t] = __builtin_amdgcn_wmma_f32_16x16x32_f16(
          false, a0, false, bt, (short)0, acc[0][t], false, false);
      acc[1][t] = __builtin_amdgcn_wmma_f32_16x16x32_f16(
          false, a1, false, bt, (short)0, acc[1][t], false, false);
    }
  }
  const int hh = (lane >> 4) & 1;
  const int c  = lane & 15;
#pragma unroll
  for (int rb = 0; rb < 2; ++rb)
#pragma unroll
    for (int t = 0; t < 4; ++t)
#pragma unroll
      for (int j = 0; j < 8; ++j)
        out[(size_t)(m0 + rb * 16 + j + 8 * hh) * kHID + n0 + t * 16 + c] =
            acc[rb][t][j];
}

// ---------------------------------------------------------------------------
// Host launcher
// ---------------------------------------------------------------------------
extern "C" void kernel_launch(void* const* d_in, const int* in_sizes, int n_in,
                              void* d_out, int out_size, void* d_ws,
                              size_t ws_size, hipStream_t stream) {
  (void)in_sizes; (void)n_in; (void)out_size; (void)ws_size;
  const float* x    = (const float*)d_in[0];
  const float* qw   = (const float*)d_in[1];
  const float* kw   = (const float*)d_in[2];
  const float* vw   = (const float*)d_in[3];
  const float* ow   = (const float*)d_in[4];
  const float* cosT = (const float*)d_in[5];
  const float* sinT = (const float*)d_in[6];
  float* out = (float*)d_out;

  // Workspace carve-out (~105 MB), 256B-aligned regions, fully rewritten
  // every call (harness does not re-poison; we never read stale data).
  char* ws = (char*)d_ws;
  size_t off = 0;
  auto carve = [&](size_t bytes) {
    char* p = ws + off;
    off += (bytes + 255) & ~(size_t)255;
    return p;
  };
  _Float16* XB = (_Float16*)carve((size_t)kTOK * kHID * 2);
  _Float16* WQ = (_Float16*)carve((size_t)kD * kHID * 2);
  _Float16* WK = (_Float16*)carve((size_t)kD * kHID * 2);
  _Float16* WV = (_Float16*)carve((size_t)kD * kHID * 2);
  _Float16* WO = (_Float16*)carve((size_t)kHID * kD * 2);
  _Float16* QB = (_Float16*)carve((size_t)kSLAB * kL * kHS * 2);
  _Float16* KB = (_Float16*)carve((size_t)kSLAB * kL * kHS * 2);
  _Float16* VB = (_Float16*)carve((size_t)kSLAB * kL * kHS * 2);
  _Float16* VT = (_Float16*)carve((size_t)kSLAB * kHS * kL * 2);
  _Float16* OB = (_Float16*)carve((size_t)kTOK * kD * 2);

  const int nx4 = kTOK * kHID / 4;   // 2,097,152
  const int nw4 = kD * kHID / 4;     //   262,144
  cvt_f32_f16_v4<<<(nx4 + 255) / 256, 256, 0, stream>>>(x,  XB, nx4);
  cvt_f32_f16_v4<<<(nw4 + 255) / 256, 256, 0, stream>>>(qw, WQ, nw4);
  cvt_f32_f16_v4<<<(nw4 + 255) / 256, 256, 0, stream>>>(kw, WK, nw4);
  cvt_f32_f16_v4<<<(nw4 + 255) / 256, 256, 0, stream>>>(vw, WV, nw4);
  cvt_f32_f16_v4<<<(nw4 + 255) / 256, 256, 0, stream>>>(ow, WO, nw4);

  // 4096 wave-tasks per GEMM (32x64 tiles), 8 waves/block -> 512 blocks.
  proj_wmma<<<512, 256, 0, stream>>>(XB, WQ, QB);
  proj_wmma<<<512, 256, 0, stream>>>(XB, WK, KB);
  proj_wmma<<<512, 256, 0, stream>>>(XB, WV, VB);

  const int nrope = kSLAB * kL * 32;   // 4,194,304 pair-threads
  rope_inplace<<<(nrope + 255) / 256, 256, 0, stream>>>(QB, cosT, sinT);
  rope_inplace<<<(nrope + 255) / 256, 256, 0, stream>>>(KB, cosT, sinT);

  const int ntr = kSLAB * kHS * kL;    // 8,388,608
  transpose_v<<<(ntr + 255) / 256, 256, 0, stream>>>(VB, VT);

  // 4096 wave-tasks (128 slabs x 32 query-blocks of 32) -> 512 blocks.
  flash_wmma<<<512, 256, 0, stream>>>(QB, KB, VT, OB);

  oproj_wmma<<<512, 256, 0, stream>>>(OB, WO, out);
}
